// VariationalGCNEncoder_32315333935771
// MI455X (gfx1250) — matrix operand order repaired and checked by reading, not compiled, not run on hardware
//
#include <hip/hip_runtime.h>

typedef __attribute__((ext_vector_type(16))) __bf16 bfvec16;
typedef __attribute__((ext_vector_type(8)))  float  v8f;

union Frag16 {
    bfvec16 v;
    uint4 q[2];
    unsigned short s[16];
};

// Branchless round-to-nearest-even f32 -> bf16. Inputs in this network are
// finite (gaussian features, ReLU activations, small weights), so we omit the
// NaN special case: it previously compiled to per-element EXEC divergence.
static __device__ __forceinline__ unsigned short f32_to_bf16(float f) {
    union { float f; unsigned int u; } c; c.f = f;
    unsigned int u = c.u;
    return (unsigned short)((u + 0x7fffu + ((u >> 16) & 1u)) >> 16);
}

static __device__ __forceinline__ void atomic_add_f32(float* p, float v) {
    __hip_atomic_fetch_add(p, v, __ATOMIC_RELAXED, __HIP_MEMORY_SCOPE_AGENT);
}

// ---------------- degree / norm ----------------
__global__ void k_deg_init(float* deg, int N) {
    int i = blockIdx.x * blockDim.x + threadIdx.x;
    if (i < N) deg[i] = 1.0f;  // self-loop
}
__global__ void k_deg_count(const int* __restrict__ dst, float* deg, int E) {
    int i = blockIdx.x * blockDim.x + threadIdx.x;
    if (i < E) atomic_add_f32(&deg[dst[i]], 1.0f);
}
__global__ void k_dinv(const float* __restrict__ deg, float* dinv, int N) {
    int i = blockIdx.x * blockDim.x + threadIdx.x;
    if (i < N) dinv[i] = rsqrtf(deg[i]);
}

// ---------------- weight prep: f32 row-major [K][N] -> bf16 N-major [N][K] ----------------
__global__ void k_prep_w1(const float* __restrict__ W, unsigned short* __restrict__ WT,
                          int K, int N) {
    int idx = blockIdx.x * blockDim.x + threadIdx.x;
    if (idx >= K * N) return;
    int k = idx / N, n = idx % N;
    WT[(size_t)n * K + k] = f32_to_bf16(W[idx]);
}
__global__ void k_prep_w2(const float* __restrict__ Wmu, const float* __restrict__ Wls,
                          unsigned short* __restrict__ WT, int K, int C) {
    int idx = blockIdx.x * blockDim.x + threadIdx.x;   // over K * 2C
    if (idx >= K * 2 * C) return;
    int k = idx / (2 * C), n = idx % (2 * C);
    float v = (n < C) ? Wmu[(size_t)k * C + n] : Wls[(size_t)k * C + (n - C)];
    WT[(size_t)n * K + k] = f32_to_bf16(v);
}

// ---------------- WMMA GEMM: C[M x 256] = A[M x K] * WT^T ----------------
// A is f32 (AF32=true, converted in-register) or bf16. WT is [256 rows][K] bf16.
// Block = 128 threads (4 waves); wave w computes columns [w*64, w*64+64).
template <bool AF32>
__global__ void __launch_bounds__(128)
k_gemm_wmma(const void* __restrict__ Aptr, const unsigned short* __restrict__ WT,
            float* __restrict__ C, int K, int ldc) {
    const int lane  = threadIdx.x & 31;
    const int wave  = threadIdx.x >> 5;
    const int mbase = blockIdx.x * 16;
    const int nbase = wave * 64;
    const int mrowA = mbase + (lane & 15);
    const int hi    = lane >> 4;           // 0 or 1
    const int koffA = hi * 8;              // A: lanes0-15 K{0..7,16..23}; lanes16-31 K{8..15,24..31}
    const int koffB = hi * 8;              // B: lanes0-15 K{0..7,16..23}; lanes16-31 K{8..15,24..31}
    const int ncol  = lane & 15;

    v8f acc0 = {0.f,0.f,0.f,0.f,0.f,0.f,0.f,0.f};
    v8f acc1 = acc0, acc2 = acc0, acc3 = acc0;

    for (int k = 0; k < K; k += 32) {
        Frag16 fa;
        if (AF32) {
            const float* ap = (const float*)Aptr + (size_t)mrowA * K + k + koffA;
            float4 a0 = *(const float4*)(ap);
            float4 a1 = *(const float4*)(ap + 4);
            float4 a2 = *(const float4*)(ap + 16);
            float4 a3 = *(const float4*)(ap + 20);
            float av[16] = {a0.x,a0.y,a0.z,a0.w, a1.x,a1.y,a1.z,a1.w,
                            a2.x,a2.y,a2.z,a2.w, a3.x,a3.y,a3.z,a3.w};
            #pragma unroll
            for (int j = 0; j < 16; ++j) fa.s[j] = f32_to_bf16(av[j]);
        } else {
            const unsigned short* ap = (const unsigned short*)Aptr + (size_t)mrowA * K + k + koffA;
            fa.q[0] = *(const uint4*)(ap);
            fa.q[1] = *(const uint4*)(ap + 16);
        }
        #pragma unroll
        for (int t = 0; t < 4; ++t) {
            Frag16 fb;
            const unsigned short* bp = WT + (size_t)(nbase + t * 16 + ncol) * K + k + koffB;
            fb.q[0] = *(const uint4*)(bp);
            fb.q[1] = *(const uint4*)(bp + 16);
            v8f& acc = (t == 0) ? acc0 : (t == 1) ? acc1 : (t == 2) ? acc2 : acc3;
            acc = __builtin_amdgcn_wmma_f32_16x16x32_bf16(
                      false, fa.v, false, fb.v, (short)0, acc, false, false);
        }
    }

    // C/D layout: lanes0-15 VGPR r -> M=r; lanes16-31 VGPR r -> M=8+r; N = lane&15
    const int mrowC = mbase + hi * 8;
    #pragma unroll
    for (int t = 0; t < 4; ++t) {
        const v8f& acc = (t == 0) ? acc0 : (t == 1) ? acc1 : (t == 2) ? acc2 : acc3;
        float* cp = C + (size_t)mrowC * ldc + nbase + t * 16 + ncol;
        #pragma unroll
        for (int r = 0; r < 8; ++r) cp[(size_t)r * ldc] = acc[r];
    }
}

// ---------------- agg init with self-loop term: agg = feat * dinv[i]^2  (F = 256) ----------
__global__ void k_self_init(const float* __restrict__ feat, const float* __restrict__ dinv,
                            float* __restrict__ agg, int total) {
    int idx = blockIdx.x * blockDim.x + threadIdx.x;
    if (idx >= total) return;
    int i = idx >> 8;
    float dv = dinv[i];
    agg[idx] = feat[idx] * dv * dv;
}

// ---------------- edge scatter-add: agg[dst] += feat[src] * dinv[src]*dinv[dst]  (F = 256) --
__global__ void __launch_bounds__(256)
k_scatter(const int* __restrict__ srcI, const int* __restrict__ dstI,
          const float* __restrict__ dinv, const float* __restrict__ feat,
          float* __restrict__ agg, int E) {
    const int lane   = threadIdx.x & 31;
    const int wave   = blockIdx.x * (blockDim.x >> 5) + (threadIdx.x >> 5);
    const int nwaves = gridDim.x * (blockDim.x >> 5);
    for (int e = wave; e < E; e += nwaves) {
        const int s = srcI[e], d = dstI[e];
        const float w = dinv[s] * dinv[d];
        const float* fs = feat + (size_t)s * 256;
        float*       od = agg  + (size_t)d * 256;
        #pragma unroll
        for (int half = 0; half < 2; ++half) {
            const int f0 = half * 128 + lane * 4;
            float4 v = *(const float4*)(fs + f0);
            atomic_add_f32(od + f0 + 0, v.x * w);
            atomic_add_f32(od + f0 + 1, v.y * w);
            atomic_add_f32(od + f0 + 2, v.z * w);
            atomic_add_f32(od + f0 + 3, v.w * w);
        }
    }
}

// ---------------- h = bf16(relu(agg + b))  (F = 256) ----------------
__global__ void k_relu_bf16(const float* __restrict__ agg, const float* __restrict__ b,
                            unsigned short* __restrict__ hbf, int total) {
    int idx = blockIdx.x * blockDim.x + threadIdx.x;
    if (idx >= total) return;
    float v = agg[idx] + b[idx & 255];
    hbf[idx] = f32_to_bf16(fmaxf(v, 0.0f));
}

// ---------------- split [mu|ls] agg + biases -> d_out ----------------
__global__ void k_final(const float* __restrict__ agg, const float* __restrict__ bmu,
                        const float* __restrict__ bls, float* __restrict__ out, int N) {
    int idx = blockIdx.x * blockDim.x + threadIdx.x;
    if (idx >= N * 256) return;
    int i = idx >> 8, f = idx & 255;
    float v = agg[idx];
    if (f < 128) out[(size_t)i * 128 + f] = v + bmu[f];
    else         out[(size_t)N * 128 + (size_t)i * 128 + (f - 128)] = v + bls[f - 128];
}

extern "C" void kernel_launch(void* const* d_in, const int* in_sizes, int n_in,
                              void* d_out, int out_size, void* d_ws, size_t ws_size,
                              hipStream_t stream) {
    const float* x   = (const float*)d_in[0];
    const int*   ei  = (const int*)d_in[1];
    const float* W1  = (const float*)d_in[2];
    const float* b1  = (const float*)d_in[3];
    const float* Wmu = (const float*)d_in[4];
    const float* bmu = (const float*)d_in[5];
    const float* Wls = (const float*)d_in[6];
    const float* bls = (const float*)d_in[7];

    const int FIN = 512, H = 256, C = 128;
    const int N = in_sizes[0] / FIN;     // 100000
    const int E = in_sizes[1] / 2;       // 3200000
    const int* srcI = ei;
    const int* dstI = ei + E;

    // workspace carve-out
    char* ws = (char*)d_ws; size_t off = 0;
    auto carve = [&](size_t bytes) -> void* {
        void* p = ws + off; off += (bytes + 255) & ~(size_t)255; return p;
    };
    unsigned short* WT1 = (unsigned short*)carve((size_t)H * FIN * 2);   // 256 x 512 bf16
    unsigned short* WT2 = (unsigned short*)carve((size_t)(2*C) * H * 2); // 256 x 256 bf16
    float* deg  = (float*)carve((size_t)N * 4);
    float* dinv = (float*)carve((size_t)N * 4);
    float* hw1  = (float*)carve((size_t)N * H * 4);   // reused as hw2
    float* agg1 = (float*)carve((size_t)N * H * 4);   // reused as agg2
    float* hw2  = hw1;
    float* agg2 = agg1;
    unsigned short* hbf = (unsigned short*)d_out;     // N*H*2 bytes <= out bytes (N*256*4)

    const int T = 256;
    const int totNH = N * H;

    // normalization
    k_deg_init <<<(N + T - 1) / T, T, 0, stream>>>(deg, N);
    k_deg_count<<<(E + T - 1) / T, T, 0, stream>>>(dstI, deg, E);
    k_dinv     <<<(N + T - 1) / T, T, 0, stream>>>(deg, dinv, N);

    // weights -> bf16, N-major
    k_prep_w1<<<(FIN * H + T - 1) / T, T, 0, stream>>>(W1, WT1, FIN, H);
    k_prep_w2<<<(H * 2 * C + T - 1) / T, T, 0, stream>>>(Wmu, Wls, WT2, H, C);

    // layer 1: hw1 = x @ W1
    k_gemm_wmma<true><<<N / 16, 128, 0, stream>>>((const void*)x, WT1, hw1, FIN, H);
    k_self_init<<<(totNH + T - 1) / T, T, 0, stream>>>(hw1, dinv, agg1, totNH);
    k_scatter  <<<4096, 256, 0, stream>>>(srcI, dstI, dinv, hw1, agg1, E);
    k_relu_bf16<<<(totNH + T - 1) / T, T, 0, stream>>>(agg1, b1, hbf, totNH);

    // layer 2 (mu | logstd fused): hw2 = h @ [Wmu | Wls]
    k_gemm_wmma<false><<<N / 16, 128, 0, stream>>>((const void*)hbf, WT2, hw2, H, 2 * C);
    k_self_init<<<(totNH + T - 1) / T, T, 0, stream>>>(hw2, dinv, agg2, totNH);
    k_scatter  <<<4096, 256, 0, stream>>>(srcI, dstI, dinv, hw2, agg2, E);
    k_final    <<<(totNH + T - 1) / T, T, 0, stream>>>(agg2, bmu, bls, (float*)d_out, N);
}
